// PureSSM_781684048752
// MI455X (gfx1250) — matrix operand order repaired and checked by reading, not compiled, not run on hardware
//
#include <hip/hip_runtime.h>
#include <hip/hip_bf16.h>
#include <math.h>

// Problem constants (from reference)
#define DMODEL 1024
#define DSTATE 64
#define DINNER 2048
#define NHEADS 32
#define HEADDIM 64
#define DCONV 4
#define DPROJ 8224          // 2*DINNER + NHEADS*(2*DSTATE+1)
#define BATCH 2
#define SEQ 1024
#define MTOT (BATCH * SEQ)  // 2048 rows

typedef float v8f __attribute__((ext_vector_type(8)));
typedef __bf16 v16bf __attribute__((ext_vector_type(16)));
typedef unsigned int v8u __attribute__((ext_vector_type(8)));

// ---- f32 -> packed bf16 pair, round-to-nearest-even ----
__device__ __forceinline__ unsigned int pack2_bf16(float lo, float hi) {
    unsigned int a = __float_as_uint(lo);
    unsigned int b = __float_as_uint(hi);
    a = (a + 0x7fffu + ((a >> 16) & 1u)) >> 16;
    b = (b + 0x7fffu + ((b >> 16) & 1u)) >> 16;
    return a | (b << 16);
}

// ---- CDNA5 async global->LDS DMA (ASYNCcnt path), inline asm ----
__device__ __forceinline__ void async_copy_b128(unsigned lds_byte_addr, const void* gaddr) {
    asm volatile("global_load_async_to_lds_b128 %0, %1, off"
                 :
                 : "v"(lds_byte_addr), "v"((unsigned long long)(uintptr_t)gaddr)
                 : "memory");
}
__device__ __forceinline__ void wait_asynccnt_4() {
    asm volatile("s_wait_asynccnt 0x4" ::: "memory");
}
__device__ __forceinline__ void wait_asynccnt_0() {
    asm volatile("s_wait_asynccnt 0x0" ::: "memory");
}

// =====================================================================
// Streaming f32 -> bf16 conversion (one-time, memory-bound).
// n must be a multiple of 8 (true for all operands here).
// =====================================================================
__global__ __launch_bounds__(256)
void cvt_f32_bf16_kernel(const float* __restrict__ src, unsigned short* __restrict__ dst,
                         long n) {
    const long i = ((long)blockIdx.x * 256 + threadIdx.x) * 8;
    if (i >= n) return;
    const float4 a = *(const float4*)(src + i);
    const float4 b = *(const float4*)(src + i + 4);
    uint4 o;
    o.x = pack2_bf16(a.x, a.y);
    o.y = pack2_bf16(a.z, a.w);
    o.z = pack2_bf16(b.x, b.y);
    o.w = pack2_bf16(b.z, b.w);
    *(uint4*)(dst + i) = o;
}

// =====================================================================
// Generic GEMM: C[M,N] = A[M,K] * B[N,K]^T   (bf16 operands, K contig)
// bf16 WMMA 16x16x32, f32 accumulate. 128x128 block tile, BK=32.
// bf16 tiles staged by async global->LDS DMA, double-buffered; hot loop
// is ds_load_b128 fragment reads + 8 WMMAs, no conversion work.
// 256 threads = 8 waves; wave grid 4(M) x 2(N); 2x4 WMMA tiles per wave.
// =====================================================================
#define BM 128
#define BN 128
#define BK 32
#define LDSW 20                // dwords per LDS row: 16 (32 bf16) + 4 pad (80B)
#define TILE_DW (BM * LDSW)    // 2560 dwords per tile buffer
// dword offsets: A0=0, A1=2560, B0=5120, B1=7680; total 10240 dwords = 40 KB

__global__ __launch_bounds__(256)
void gemm_bf16_wmma(const unsigned short* __restrict__ A,
                    const unsigned short* __restrict__ Bw,
                    float* __restrict__ C, int M, int N, int K) {
    extern __shared__ unsigned int smem[];

    const int tid  = threadIdx.x;
    const int lane = tid & 31;
    const int wv   = tid >> 5;     // 0..7
    const int wr   = wv >> 1;      // 0..3 -> M offset wr*32
    const int wc   = wv & 1;       // 0..1 -> N offset wc*64

    const int mBase = blockIdx.y * BM;
    const int nBase = blockIdx.x * BN;

    const int srow  = tid >> 1;    // staging: row 0..127
    const int shalf = tid & 1;     // staging: bf16 elems [shalf*16, +16)

    const int lm    = lane & 15;          // row/col within 16x16 tile
    const int koffA = (lane >> 4) * 8;    // A frag dword offset (0 or 8 = 16 bf16)
    const int koffB = (lane >> 4) * 8;    // B frag dword offset (0 or 8 = 16 bf16)
    const int rsel  = (lane >> 4) * 8;    // C/D: lanes 16-31 hold M+8

    // Per-thread staging bases (LDS byte addresses for async DMA)
    const unsigned ldsA0 = (unsigned)(uintptr_t)(const void*)(smem + srow * LDSW + shalf * 8);
    const unsigned ldsB0 = ldsA0 + 2u * TILE_DW * 4u;   // B region is 2 tiles after A
    const int gmRow = mBase + srow;
    const int gnRow = nBase + srow;

    v8f acc[2][4];
    #pragma unroll
    for (int mi = 0; mi < 2; mi++)
        #pragma unroll
        for (int ni = 0; ni < 4; ni++)
            #pragma unroll
            for (int e = 0; e < 8; e++) acc[mi][ni][e] = 0.0f;

    // ---- issue async DMA of one K-tile (bf16) into buffer `buf` ----
    auto issue_tile = [&](int buf, int kk) {
        const unsigned boff = (unsigned)buf * TILE_DW * 4u;
        if (gmRow < M) {
            const unsigned short* g = A + (size_t)gmRow * K + kk + shalf * 16;
            const unsigned l = ldsA0 + boff;
            async_copy_b128(l +  0, g + 0);   // 8 bf16 = 16B
            async_copy_b128(l + 16, g + 8);
        }
        if (gnRow < N) {
            const unsigned short* g = Bw + (size_t)gnRow * K + kk + shalf * 16;
            const unsigned l = ldsB0 + boff;
            async_copy_b128(l +  0, g + 0);
            async_copy_b128(l + 16, g + 8);
        }
    };

    issue_tile(0, 0);   // prologue

    const int nsteps = K / BK;
    for (int step = 0; step < nsteps; step++) {
        const int cur = step & 1;
        const bool has_next = (step + 1) < nsteps;
        if (has_next) issue_tile(cur ^ 1, (step + 1) * BK);

        // current buffer's 4 loads are the oldest outstanding; newest 4 are next's
        if (has_next) wait_asynccnt_4(); else wait_asynccnt_0();
        __syncthreads();

        const unsigned int* Acur = smem + cur * TILE_DW;
        const unsigned int* Bcur = smem + 2 * TILE_DW + cur * TILE_DW;

        // ---- fragment reads (merge into ds_load_b128) + 8 WMMAs ----
        v16bf afrag[2], bfrag[4];
        #pragma unroll
        for (int mi = 0; mi < 2; mi++) {
            const unsigned int* pr = Acur + (wr * 32 + mi * 16 + lm) * LDSW;
            v8u r;
            r[0] = pr[koffA + 0]; r[1] = pr[koffA + 1];
            r[2] = pr[koffA + 2]; r[3] = pr[koffA + 3];
            r[4] = pr[koffA + 4]; r[5] = pr[koffA + 5];
            r[6] = pr[koffA + 6]; r[7] = pr[koffA + 7];
            afrag[mi] = __builtin_bit_cast(v16bf, r);
        }
        #pragma unroll
        for (int ni = 0; ni < 4; ni++) {
            const unsigned int* pb = Bcur + (wc * 64 + ni * 16 + lm) * LDSW;
            v8u r;
            #pragma unroll
            for (int j = 0; j < 8; j++) r[j] = pb[koffB + j];
            bfrag[ni] = __builtin_bit_cast(v16bf, r);
        }
        #pragma unroll
        for (int mi = 0; mi < 2; mi++)
            #pragma unroll
            for (int ni = 0; ni < 4; ni++)
                acc[mi][ni] = __builtin_amdgcn_wmma_f32_16x16x32_bf16(
                    false, afrag[mi], false, bfrag[ni],
                    (short)0, acc[mi][ni], false, false);

        __syncthreads();   // all reads of `cur` done before next iter's DMA overwrites it
    }

    // ---- epilogue (C/D layout: VGPR v -> M = v + 8*(lane>=16), N = lane%16) ----
    #pragma unroll
    for (int mi = 0; mi < 2; mi++) {
        #pragma unroll
        for (int ni = 0; ni < 4; ni++) {
            const int col   = nBase + wc * 64 + ni * 16 + lm;
            const int rbase = mBase + wr * 32 + mi * 16 + rsel;
            if (col < N) {
                #pragma unroll
                for (int v = 0; v < 8; v++) {
                    const int r = rbase + v;
                    if (r < M) C[(size_t)r * N + col] = acc[mi][ni][v];
                }
            }
        }
    }
}

// =====================================================================
// Depthwise causal conv (width 4) + bias + SiLU over x_inner slice of proj
// =====================================================================
__global__ __launch_bounds__(256)
void conv_silu_kernel(const float* __restrict__ proj, const float* __restrict__ cw,
                      const float* __restrict__ cb, float* __restrict__ xconv) {
    const int idx = blockIdx.x * 256 + threadIdx.x;
    if (idx >= MTOT * DINNER) return;
    const int c = idx & (DINNER - 1);
    const int m = idx >> 11;           // DINNER == 2048
    const int t = m & (SEQ - 1);

    float accv = cb[c];
    #pragma unroll
    for (int j = 0; j < DCONV; j++) {
        const int back = (DCONV - 1) - j;
        if (t - back >= 0)
            accv += cw[c * DCONV + j] * proj[(size_t)(m - back) * DPROJ + c];
    }
    accv = accv / (1.0f + __expf(-accv));   // SiLU
    xconv[idx] = accv;
}

// =====================================================================
// Selective scan: one block per (batch, head). 256 threads.
// Thread (sg, d): owns h[s = sg*16+i][d], i = 0..15, in registers.
// Fuses +D*x and *silu(z); emits bf16 directly (feeds GEMM2's A operand).
// =====================================================================
__global__ __launch_bounds__(256)
void ssm_scan_kernel(const float* __restrict__ proj, const float* __restrict__ xconv,
                     const float* __restrict__ A_log, const float* __restrict__ Dp,
                     const float* __restrict__ dt_bias, unsigned short* __restrict__ ybf) {
    const int b   = blockIdx.x >> 5;   // NHEADS == 32
    const int n   = blockIdx.x & 31;
    const int tid = threadIdx.x;
    const int d   = tid & 63;
    const int sg  = tid >> 6;          // 0..3

    __shared__ float sA[DSTATE], sdA[DSTATE], sdB[DSTATE], sC[DSTATE];
    __shared__ float sx[HEADDIM], part[256];

    if (tid < DSTATE) sA[tid] = -__expf(A_log[n * DSTATE + tid]);
    const float dtb = dt_bias[n];
    const float Dn  = Dp[n];

    float h[16];
    #pragma unroll
    for (int i = 0; i < 16; i++) h[i] = 0.0f;

    __syncthreads();

    const int rowBase = b * SEQ;
    for (int t = 0; t < SEQ; t++) {
        const size_t base = (size_t)(rowBase + t) * DPROJ;
        if (tid < DSTATE) {
            const float* bc = proj + base + 2 * DINNER + n * (2 * DSTATE + 1);
            const float dtraw = bc[2 * DSTATE] + dtb;
            const float dtv = (dtraw > 20.0f) ? dtraw : log1pf(__expf(dtraw));  // softplus
            sdA[tid] = __expf(sA[tid] * dtv);
            sdB[tid] = bc[tid] * dtv;
            sC[tid]  = bc[DSTATE + tid];
        } else if (tid < DSTATE + HEADDIM) {
            const int dd = tid - DSTATE;
            sx[dd] = xconv[(size_t)(rowBase + t) * DINNER + n * HEADDIM + dd];
        }
        __syncthreads();

        const float xd = sx[d];
        float accv = 0.0f;
        #pragma unroll
        for (int i = 0; i < 16; i++) {
            const int s = sg * 16 + i;
            h[i] = sdA[s] * h[i] + sdB[s] * xd;
            accv += sC[s] * h[i];
        }
        part[tid] = accv;
        __syncthreads();

        if (sg == 0) {
            float y = part[d] + part[64 + d] + part[128 + d] + part[192 + d] + Dn * xd;
            const float zv = proj[base + DINNER + n * HEADDIM + d];
            const float sz = zv / (1.0f + __expf(-zv));  // silu(z)
            const float o  = y * sz;
            ybf[(size_t)(rowBase + t) * DINNER + n * HEADDIM + d] =
                (unsigned short)(pack2_bf16(o, 0.0f) & 0xffffu);
        }
    }
}

// =====================================================================
// Launch: cvt(x, W_in, W_out) -> GEMM1 -> conv+SiLU -> scan(bf16 out) -> GEMM2
// =====================================================================
extern "C" void kernel_launch(void* const* d_in, const int* in_sizes, int n_in,
                              void* d_out, int out_size, void* d_ws, size_t ws_size,
                              hipStream_t stream) {
    const float* x       = (const float*)d_in[0];
    const float* W_in    = (const float*)d_in[1];
    const float* conv_w  = (const float*)d_in[2];
    const float* conv_b  = (const float*)d_in[3];
    const float* A_log   = (const float*)d_in[4];
    const float* Dp      = (const float*)d_in[5];
    const float* dt_bias = (const float*)d_in[6];
    const float* W_out   = (const float*)d_in[7];
    float* out = (float*)d_out;

    // ---- workspace layout ----
    float* proj  = (float*)d_ws;                       // MTOT x DPROJ           (f32)
    float* xconv = proj  + (size_t)MTOT * DPROJ;       // MTOT x DINNER          (f32)
    unsigned short* bfbase = (unsigned short*)(xconv + (size_t)MTOT * DINNER);
    unsigned short* xbf    = bfbase;                                   // MTOT x DMODEL
    unsigned short* Winbf  = xbf   + (size_t)MTOT * DMODEL;            // DPROJ x DMODEL
    unsigned short* ybf    = Winbf + (size_t)DPROJ * DMODEL;           // MTOT x DINNER
    unsigned short* Woutbf = ybf   + (size_t)MTOT * DINNER;            // DMODEL x DINNER

    const size_t gemm_lds = 4u * TILE_DW * 4u;   // 40 KB double-buffered bf16 tiles

    // 0) one-time f32 -> bf16 operand conversion (streaming, ~4 us at 23.3 TB/s)
    {
        const long nx = (long)MTOT * DMODEL;       // 2,097,152
        const long nw = (long)DPROJ * DMODEL;      // 8,421,376
        const long no = (long)DMODEL * DINNER;     // 2,097,152
        cvt_f32_bf16_kernel<<<(int)((nx / 8 + 255) / 256), 256, 0, stream>>>(x, xbf, nx);
        cvt_f32_bf16_kernel<<<(int)((nw / 8 + 255) / 256), 256, 0, stream>>>(W_in, Winbf, nw);
        cvt_f32_bf16_kernel<<<(int)((no / 8 + 255) / 256), 256, 0, stream>>>(W_out, Woutbf, no);
    }
    // 1) proj = x @ W_in^T   (M=2048, N=8224, K=1024)
    {
        dim3 grid((DPROJ + BN - 1) / BN, (MTOT + BM - 1) / BM);
        gemm_bf16_wmma<<<grid, 256, gemm_lds, stream>>>(xbf, Winbf, proj, MTOT, DPROJ, DMODEL);
    }
    // 2) depthwise conv + SiLU
    {
        const int total = MTOT * DINNER;
        conv_silu_kernel<<<(total + 255) / 256, 256, 0, stream>>>(proj, conv_w, conv_b, xconv);
    }
    // 3) selective scan (fused +D*x and *silu(z)), bf16 output
    {
        ssm_scan_kernel<<<BATCH * NHEADS, 256, 0, stream>>>(proj, xconv, A_log, Dp,
                                                            dt_bias, ybf);
    }
    // 4) out = y @ W_out^T  (M=2048, N=1024, K=2048)
    {
        dim3 grid((DMODEL + BN - 1) / BN, (MTOT + BM - 1) / BM);
        gemm_bf16_wmma<<<grid, 256, gemm_lds, stream>>>(ybf, Woutbf, out, MTOT, DMODEL, DINNER);
    }
}
